// Attention_10222022164791
// MI455X (gfx1250) — matrix operand order repaired and checked by reading, not compiled, not run on hardware
//
#include <hip/hip_runtime.h>
#include <hip/hip_bf16.h>

// SAGAN self-attention block, flash-attention style fusion for gfx1250 (wave32, WMMA).
//   B=4, C=64, H=W=128 -> N=16384 queries, M=4096 pooled kv positions.
//   scores/PV matmuls via v_wmma_f32_16x16x32_f16; softmax is online (never materialize [N,M]).
//   theta is pre-scaled by log2(e) so softmax uses raw v_exp_f32 (exp2) with no per-score mul.

typedef __attribute__((ext_vector_type(16))) _Float16 v16h;
typedef __attribute__((ext_vector_type(8)))  float    v8f;

#define CH   64
#define Hh   128
#define Ww   128
#define NN   (Hh*Ww)   // 16384
#define MM   (NN/4)    // 4096
#define BB   4
#define LOG2E 1.4426950408889634f

union U4H8 { uint4 u; _Float16 h[8]; };

// ---------------- prep: theta = log2e * (w_theta (8x64) @ x), f16 as thetaT[b][n][8] ------------
__global__ __launch_bounds__(256) void k_theta(const float* __restrict__ x,
                                               const float* __restrict__ w_theta,
                                               _Float16* __restrict__ thetaT) {
  __shared__ float ws[8*CH];
  for (int i = threadIdx.x; i < 8*CH; i += 256) ws[i] = w_theta[i];
  __syncthreads();
  int gid = blockIdx.x * 256 + threadIdx.x;   // over B*N
  int b = gid >> 14, n = gid & (NN-1);
  const float* xb = x + (size_t)b * CH * NN + n;
  float acc[8];
#pragma unroll
  for (int c = 0; c < 8; ++c) acc[c] = 0.f;
  for (int ci = 0; ci < CH; ++ci) {
    float xv = xb[(size_t)ci * NN];
#pragma unroll
    for (int c = 0; c < 8; ++c) acc[c] += ws[c*CH + ci] * xv;
  }
  U4H8 r;
#pragma unroll
  for (int c = 0; c < 8; ++c) r.h[c] = (_Float16)(acc[c] * LOG2E);  // fold exp2 base conversion
  *reinterpret_cast<uint4*>(thetaT + (size_t)gid * 8) = r.u;
}

// ------------- prep: phi / g = maxpool2x2(conv1x1(x)); phiT[b][m][8], gmat[b][32][M] -------------
__global__ __launch_bounds__(128) void k_phig(const float* __restrict__ x,
                                              const float* __restrict__ w_phi,
                                              const float* __restrict__ w_g,
                                              _Float16* __restrict__ phiT,
                                              _Float16* __restrict__ gmat) {
  __shared__ float wp[8*CH];
  __shared__ float wg[32*CH];
  for (int i = threadIdx.x; i < 8*CH;  i += 128) wp[i] = w_phi[i];
  for (int i = threadIdx.x; i < 32*CH; i += 128) wg[i] = w_g[i];
  __syncthreads();
  int gid = blockIdx.x * 128 + threadIdx.x;   // over B*M
  int b = gid >> 12, m = gid & (MM-1);
  int h2 = m >> 6, w2 = m & 63;               // pooled grid 64x64
  int n00 = (h2*2)*Ww + w2*2;
  const int off[4] = {0, 1, Ww, Ww+1};
  const float* xb = x + (size_t)b * CH * NN;
  float pm[8], gm[32];
#pragma unroll
  for (int c = 0; c < 8;  ++c) pm[c] = -1e30f;
#pragma unroll
  for (int c = 0; c < 32; ++c) gm[c] = -1e30f;
  for (int p = 0; p < 4; ++p) {
    const float* xp = xb + n00 + off[p];
    float pa[8], ga[32];
#pragma unroll
    for (int c = 0; c < 8;  ++c) pa[c] = 0.f;
#pragma unroll
    for (int c = 0; c < 32; ++c) ga[c] = 0.f;
    for (int ci = 0; ci < CH; ++ci) {
      float xv = xp[(size_t)ci * NN];
#pragma unroll
      for (int c = 0; c < 8;  ++c) pa[c] += wp[c*CH + ci] * xv;
#pragma unroll
      for (int c = 0; c < 32; ++c) ga[c] += wg[c*CH + ci] * xv;
    }
#pragma unroll
    for (int c = 0; c < 8;  ++c) pm[c] = fmaxf(pm[c], pa[c]);
#pragma unroll
    for (int c = 0; c < 32; ++c) gm[c] = fmaxf(gm[c], ga[c]);
  }
  U4H8 r;
#pragma unroll
  for (int c = 0; c < 8; ++c) r.h[c] = (_Float16)pm[c];
  *reinterpret_cast<uint4*>(phiT + (size_t)gid * 8) = r.u;
#pragma unroll
  for (int c = 0; c < 32; ++c)
    gmat[((size_t)b*32 + c)*MM + m] = (_Float16)gm[c];
}

// ---------------- fused flash attention + output proj, 1 wave per 16-query block ----------------
// scores D layout: rows = keys (VGPRs), cols = queries (lanes) -> per-lane softmax over keys.
// exp'd scores match the f16 A-matrix 16x32 layout exactly (in-lane) for the PV WMMA.
// 64 keys per iteration: softmax cross-lane traffic amortized over 4 score tiles.
__global__ __launch_bounds__(32) void k_attn(const float* __restrict__ x,
                                             const float* __restrict__ w_o,
                                             const float* __restrict__ gamma_p,
                                             const _Float16* __restrict__ thetaT,
                                             const _Float16* __restrict__ phiT,
                                             const _Float16* __restrict__ gmat,
                                             float* __restrict__ out) {
  __shared__ float om[32*16];                 // o_mid transpose staging [ci][q]
  int blk  = blockIdx.x;
  int b    = blk >> 10;                       // 1024 query-blocks per batch
  int qb   = (blk & 1023) << 4;
  int lane = threadIdx.x & 31;
  int lq   = lane & 15;
  int hi   = lane >> 4;

  // B operand (theta): 32x16, lanes 0-15 hold K=0..15 (only K=0..7 nonzero = channels)
  v16h Bth;
#pragma unroll
  for (int i = 0; i < 16; ++i) Bth[i] = (_Float16)0.f;
  if (hi == 0) {
    U4H8 t; t.u = *reinterpret_cast<const uint4*>(thetaT + ((size_t)b*NN + qb + lq)*8);
#pragma unroll
    for (int i = 0; i < 8; ++i) Bth[i] = t.h[i];
  }

  const _Float16* phib = phiT + (size_t)b * MM * 8;
  const _Float16* g1   = gmat + ((size_t)b*32 + lq)      * MM;  // channels 0..15
  const _Float16* g2   = gmat + ((size_t)b*32 + 16 + lq) * MM;  // channels 16..31

  v8f acc0, acc1, vzero;
#pragma unroll
  for (int i = 0; i < 8; ++i) { acc0[i] = 0.f; acc1[i] = 0.f; vzero[i] = 0.f; }
  float mrow = -1e30f, lrow = 0.f;

  for (int kc = 0; kc < MM; kc += 64) {
    // prefetch next chunk of phi records + this wave's two g rows (global_prefetch_b8)
    if (kc + 64 < MM) {
      __builtin_prefetch(phib + (size_t)(kc + 64 + lane)*8, 0, 1);
      __builtin_prefetch(g1 + kc + 64, 0, 1);
      __builtin_prefetch(g2 + kc + 64, 0, 1);
    }

    // A operands (phi): 4 tiles of 16 keys, loaded directly by lanes 0-15 (K=0..7 = channels)
    v16h A0, A1, A2, A3;
#pragma unroll
    for (int i = 0; i < 16; ++i) {
      A0[i] = (_Float16)0.f; A1[i] = (_Float16)0.f;
      A2[i] = (_Float16)0.f; A3[i] = (_Float16)0.f;
    }
    if (hi == 0) {
      U4H8 p0, p1, p2, p3;
      p0.u = *reinterpret_cast<const uint4*>(phib + (size_t)(kc      + lq)*8);
      p1.u = *reinterpret_cast<const uint4*>(phib + (size_t)(kc + 16 + lq)*8);
      p2.u = *reinterpret_cast<const uint4*>(phib + (size_t)(kc + 32 + lq)*8);
      p3.u = *reinterpret_cast<const uint4*>(phib + (size_t)(kc + 48 + lq)*8);
#pragma unroll
      for (int i = 0; i < 8; ++i) {
        A0[i] = p0.h[i]; A1[i] = p1.h[i]; A2[i] = p2.h[i]; A3[i] = p3.h[i];
      }
    }
    // scores (already in log2-domain): D[key, query], 4 tiles of 16 keys
    v8f S0 = __builtin_amdgcn_wmma_f32_16x16x32_f16(false, A0, false, Bth, (short)0, vzero, false, false);
    v8f S1 = __builtin_amdgcn_wmma_f32_16x16x32_f16(false, A1, false, Bth, (short)0, vzero, false, false);
    v8f S2 = __builtin_amdgcn_wmma_f32_16x16x32_f16(false, A2, false, Bth, (short)0, vzero, false, false);
    v8f S3 = __builtin_amdgcn_wmma_f32_16x16x32_f16(false, A3, false, Bth, (short)0, vzero, false, false);

    // online softmax over 64 keys (per query = per lane column; combine halves via shfl_xor 16)
    float cmax = -1e30f;
#pragma unroll
    for (int i = 0; i < 8; ++i) {
      cmax = fmaxf(cmax, fmaxf(S0[i], S1[i]));
      cmax = fmaxf(cmax, fmaxf(S2[i], S3[i]));
    }
    cmax = fmaxf(cmax, __shfl_xor(cmax, 16));
    float mnew  = fmaxf(mrow, cmax);
    float alpha = __builtin_amdgcn_exp2f(mrow - mnew);
    float p0[8], p1[8], p2[8], p3[8], ls = 0.f;
#pragma unroll
    for (int i = 0; i < 8; ++i) {
      p0[i] = __builtin_amdgcn_exp2f(S0[i] - mnew);
      p1[i] = __builtin_amdgcn_exp2f(S1[i] - mnew);
      p2[i] = __builtin_amdgcn_exp2f(S2[i] - mnew);
      p3[i] = __builtin_amdgcn_exp2f(S3[i] - mnew);
      ls += (p0[i] + p1[i]) + (p2[i] + p3[i]);
    }
    ls += __shfl_xor(ls, 16);
    lrow = lrow * alpha + ls;
    mrow = mnew;

    // PV A operands: in-lane repack of exp'd scores (matches 16-bit A 16x32 layout)
    v16h Ap0, Ap1;
#pragma unroll
    for (int i = 0; i < 8; ++i) {
      Ap0[i] = (_Float16)p0[i]; Ap0[8+i] = (_Float16)p1[i];
      Ap1[i] = (_Float16)p2[i]; Ap1[8+i] = (_Float16)p3[i];
    }

    // rescale accumulators by per-query alpha (acc rows = queries r+8*hi), once per 64 keys
#pragma unroll
    for (int r = 0; r < 8; ++r) {
      float af = __shfl(alpha, r + 8*hi);
      acc0[r] *= af; acc1[r] *= af;
    }

    // PV B operands: g^T 32x16; lane = channel col, K rows = keys 8hi..8hi+7 and 16+8hi..16+8hi+7 per half
    const _Float16* g1p = g1 + kc + 8*hi;
    const _Float16* g2p = g2 + kc + 8*hi;
    U4H8 ba0, ba1, bb0, bb1, bc0, bc1, bd0, bd1;
    ba0.u = *reinterpret_cast<const uint4*>(g1p);          // grp0 ch0-15
    ba1.u = *reinterpret_cast<const uint4*>(g1p + 16);
    bb0.u = *reinterpret_cast<const uint4*>(g2p);          // grp0 ch16-31
    bb1.u = *reinterpret_cast<const uint4*>(g2p + 16);
    bc0.u = *reinterpret_cast<const uint4*>(g1p + 32);     // grp1 ch0-15
    bc1.u = *reinterpret_cast<const uint4*>(g1p + 48);
    bd0.u = *reinterpret_cast<const uint4*>(g2p + 32);     // grp1 ch16-31
    bd1.u = *reinterpret_cast<const uint4*>(g2p + 48);
    v16h Bg0a, Bg0b, Bg1a, Bg1b;
#pragma unroll
    for (int i = 0; i < 8; ++i) {
      Bg0a[i] = ba0.h[i]; Bg0a[8+i] = ba1.h[i];
      Bg0b[i] = bb0.h[i]; Bg0b[8+i] = bb1.h[i];
      Bg1a[i] = bc0.h[i]; Bg1a[8+i] = bc1.h[i];
      Bg1b[i] = bd0.h[i]; Bg1b[8+i] = bd1.h[i];
    }
    acc0 = __builtin_amdgcn_wmma_f32_16x16x32_f16(false, Ap0, false, Bg0a, (short)0, acc0, false, false);
    acc1 = __builtin_amdgcn_wmma_f32_16x16x32_f16(false, Ap0, false, Bg0b, (short)0, acc1, false, false);
    acc0 = __builtin_amdgcn_wmma_f32_16x16x32_f16(false, Ap1, false, Bg1a, (short)0, acc0, false, false);
    acc1 = __builtin_amdgcn_wmma_f32_16x16x32_f16(false, Ap1, false, Bg1b, (short)0, acc1, false, false);
  }

  // normalize by row sums, transpose o_mid [32ci x 16q] through LDS
#pragma unroll
  for (int r = 0; r < 8; ++r) {
    float li = 1.f / __shfl(lrow, r + 8*hi);
    om[lq*16      + (r + 8*hi)] = acc0[r] * li;
    om[(16+lq)*16 + (r + 8*hi)] = acc1[r] * li;
  }
  __syncthreads();
  // B operand for output proj: K=ci (32), N=queries
  v16h Bom;
#pragma unroll
  for (int e = 0; e < 16; ++e) Bom[e] = (_Float16)om[((e & 7) + 8*hi + 16*(e >> 3))*16 + lq];

  float gamma = gamma_p[0];
  const float* xb = x   + (size_t)b * CH * NN;
  float*       ob = out + (size_t)b * CH * NN;
#pragma unroll
  for (int t = 0; t < 4; ++t) {
    // A operand: w_o rows 16t..16t+15, K=ci split per lane half (0..7/16..23 vs 8..15/24..31)
    const float* wr = w_o + (t*16 + lq)*32;
    v16h Aw;
#pragma unroll
    for (int e = 0; e < 8; ++e) {
      Aw[e]   = (_Float16)wr[8*hi + e];
      Aw[8+e] = (_Float16)wr[16 + 8*hi + e];
    }
    v8f D = __builtin_amdgcn_wmma_f32_16x16x32_f16(false, Aw, false, Bom, (short)0, vzero, false, false);
#pragma unroll
    for (int r = 0; r < 8; ++r) {
      int c = t*16 + r + 8*hi;
      size_t idx = (size_t)c*NN + qb + lq;
      ob[idx] = gamma * D[r] + xb[idx];
    }
  }
}

extern "C" void kernel_launch(void* const* d_in, const int* in_sizes, int n_in,
                              void* d_out, int out_size, void* d_ws, size_t ws_size,
                              hipStream_t stream) {
  const float* x       = (const float*)d_in[0];
  const float* w_theta = (const float*)d_in[1];
  const float* w_phi   = (const float*)d_in[2];
  const float* w_g     = (const float*)d_in[3];
  const float* w_o     = (const float*)d_in[4];
  const float* gamma   = (const float*)d_in[5];
  float* out = (float*)d_out;

  char* ws = (char*)d_ws;
  size_t off0 = 0;                                  // thetaT: B*N*8 f16 = 1 MB
  size_t off1 = off0 + (size_t)BB*NN*8*2;           // phiT:   B*M*8 f16 = 256 KB
  size_t off2 = off1 + (size_t)BB*MM*8*2;           // gmat:   B*32*M f16 = 1 MB
  _Float16* thetaT = (_Float16*)(ws + off0);
  _Float16* phiT   = (_Float16*)(ws + off1);
  _Float16* gmat   = (_Float16*)(ws + off2);

  k_theta<<<(BB*NN)/256, 256, 0, stream>>>(x, w_theta, thetaT);
  k_phig <<<(BB*MM)/128, 128, 0, stream>>>(x, w_phi, w_g, phiT, gmat);
  k_attn <<<BB*(NN/16), 32, 0, stream>>>(x, w_o, gamma, thetaT, phiT, gmat, out);
}
